// ChebyKANLinear_20942260536114
// MI455X (gfx1250) — matrix operand order repaired and checked by reading, not compiled, not run on hardware
//
#include <hip/hip_runtime.h>
#include <hip/hip_bf16.h>

// ChebyKAN linear, restructured as ONE f16 WMMA GEMM:
//   out[b,o] = sum_k F[b,k] * G[o,k] + bias[o],  K = 1024*12 = 12288
// F[b, 12i+j] = { silu(x[b,i]), T_0..T_10(tanh(clip(x[b,i]))) }   (f16)
// G[o, 12i+j] = { base_weight[o,i], cheby_coeffs[o,i,0..10] }      (f16)
// f32 accumulation via v_wmma_f32_16x16x32_f16 (wave32, gfx1250).
// Global->LDS staging uses CDNA5 async copies (GLOBAL_LOAD_ASYNC_TO_LDS_B128,
// tracked by ASYNCcnt) so no staging VGPRs are burned.

typedef _Float16 half_t;
typedef __attribute__((ext_vector_type(8)))  _Float16 v8h;
typedef __attribute__((ext_vector_type(16))) _Float16 v16h;
typedef __attribute__((ext_vector_type(8)))  float    v8f;

#define IN_F    1024
#define OUT_F   1024
#define DEG     10
#define KPER    (DEG + 2)          // 12 features per input channel
#define KDIM    (IN_F * KPER)      // 12288
#define BM      256
#define BN      128
#define BK      32
#define LDT     40                 // padded LDS row stride in halves (20 dwords)

// ---------------------------------------------------------------------------
// CDNA5 async global->LDS copy (ASYNCcnt path), per CDNA5 ISA
// ---------------------------------------------------------------------------
__device__ __forceinline__ void async_load_b128(unsigned lds_byte_addr,
                                                const void* gaddr) {
  asm volatile("global_load_async_to_lds_b128 %0, %1, off"
               :: "v"(lds_byte_addr), "v"(gaddr)
               : "memory");
}
__device__ __forceinline__ void wait_async0() {
  asm volatile("s_wait_asynccnt 0" ::: "memory");
}
// Low 32 bits of a flat LDS address == wave-relative LDS byte offset.
__device__ __forceinline__ unsigned lds_off(const void* p) {
  return (unsigned)(size_t)p;
}

// ---------------------------------------------------------------------------
// Kernel 1: per-element feature expansion (SiLU + Chebyshev recurrence).
// cos(n*acos(t)) == T_n(t) via T_{n+1} = 2 t T_n - T_{n-1} (no acos/cos).
// ---------------------------------------------------------------------------
__global__ __launch_bounds__(256) void cheby_expand_kernel(
    const float* __restrict__ x, half_t* __restrict__ F, int total)
{
  int idx = blockIdx.x * 256 + threadIdx.x;
  if (idx >= total) return;
  float v  = x[idx];
  float s  = v / (1.0f + __expf(-v));            // SiLU
  float vc = fminf(fmaxf(v, -1.0f), 1.0f);
  float xc = tanhf(vc);
  union { half_t h[KPER]; unsigned long long q[3]; } u;
  u.h[0] = (half_t)s;
  u.h[1] = (half_t)1.0f;                         // T_0
  u.h[2] = (half_t)xc;                           // T_1
  float tkm1 = 1.0f, tk = xc;
  #pragma unroll
  for (int n = 2; n <= DEG; ++n) {
    float t = 2.0f * xc * tk - tkm1;
    tkm1 = tk; tk = t;
    u.h[n + 1] = (half_t)t;
  }
  unsigned long long* dst = (unsigned long long*)(F + (size_t)idx * KPER);
  dst[0] = u.q[0]; dst[1] = u.q[1]; dst[2] = u.q[2];
}

// ---------------------------------------------------------------------------
// Kernel 2: pack base_weight + cheby_coeffs into G[o, 12i+j] as f16.
// ---------------------------------------------------------------------------
__global__ __launch_bounds__(256) void cheby_pack_weights_kernel(
    const float* __restrict__ w, const float* __restrict__ cheb,
    half_t* __restrict__ G)
{
  int idx = blockIdx.x * 256 + threadIdx.x;      // 0 .. OUT_F*IN_F-1
  int o = idx >> 10;
  int i = idx & (IN_F - 1);
  const float* cp = cheb + ((size_t)o * IN_F + i) * (DEG + 1);
  union { half_t h[KPER]; unsigned long long q[3]; } u;
  u.h[0] = (half_t)w[(size_t)o * IN_F + i];
  #pragma unroll
  for (int k = 0; k <= DEG; ++k) u.h[k + 1] = (half_t)cp[k];
  unsigned long long* dst =
      (unsigned long long*)(G + (size_t)o * KDIM + (size_t)i * KPER);
  dst[0] = u.q[0]; dst[1] = u.q[1]; dst[2] = u.q[2];
}

// ---------------------------------------------------------------------------
// Kernel 3: NT GEMM  C[M,1024] = F[M,K] * G[1024,K]^T + bias
// 256 threads = 8 waves (4x2); block tile 256x128; wave tile 64x64 (4x4 WMMA).
// Double-buffered LDS fed by async global->LDS b128 copies.
// Per k-step: 16 ds_load_b128 feed 16 v_wmma (1:1 ratio).
// ---------------------------------------------------------------------------
__global__ __launch_bounds__(256) void cheby_gemm_kernel(
    const half_t* __restrict__ A, const half_t* __restrict__ Bm,
    const float* __restrict__ bias, float* __restrict__ C, int M)
{
  __shared__ __align__(16) half_t sA[2][BM * LDT];   // 2 x 20480 B
  __shared__ __align__(16) half_t sB[2][BN * LDT];   // 2 x 10240 B

  const int tid  = threadIdx.x;
  const int lane = tid & 31;
  const int wid  = tid >> 5;
  const int wm   = wid & 3;          // 4 waves of 64 rows
  const int wn   = wid >> 2;         // 2 waves of 64 cols
  const int bm   = blockIdx.x * BM;
  const int bn   = blockIdx.y * BN;

  // Async staging: thread owns rows (tid>>2)+{0,64,128,192} of A (4 x b128)
  // and rows (tid>>2)+{0,64} of B (2 x b128), column chunk (tid&3)*8 halves.
  const int grow = tid >> 2;         // 0..63
  const int gcol = (tid & 3) * 8;
  const half_t* gA = A  + (size_t)(bm + grow) * KDIM + gcol;
  const half_t* gB = Bm + (size_t)(bn + grow) * KDIM + gcol;
  const unsigned ldsA0 = lds_off(&sA[0][grow * LDT + gcol]);
  const unsigned ldsB0 = lds_off(&sB[0][grow * LDT + gcol]);
  const unsigned bufA  = BM * LDT * 2;        // bytes per A buffer (20480)
  const unsigned bufB  = BN * LDT * 2;        // bytes per B buffer (10240)
  const unsigned rstep = 64 * LDT * 2;        // 64-row LDS stride (5120)

  v8f zero = {};
  v8f acc[4][4];
  #pragma unroll
  for (int i = 0; i < 4; ++i)
    #pragma unroll
    for (int j = 0; j < 4; ++j) acc[i][j] = zero;

  // prologue: async-load k-tile 0 into buffer 0
  #pragma unroll
  for (int j = 0; j < 4; ++j)
    async_load_b128(ldsA0 + j * rstep, gA + (size_t)(j * 64) * KDIM);
  #pragma unroll
  for (int j = 0; j < 2; ++j)
    async_load_b128(ldsB0 + j * rstep, gB + (size_t)(j * 64) * KDIM);
  wait_async0();
  __syncthreads();

  const int NK = KDIM / BK;          // 384 k-tiles
  for (int kt = 0; kt < NK; ++kt) {
    const int cur = kt & 1;
    if (kt + 1 < NK) {               // async prefetch next tile (other buffer)
      const size_t koff = (size_t)(kt + 1) * BK;
      const unsigned nb = (unsigned)(cur ^ 1);
      #pragma unroll
      for (int j = 0; j < 4; ++j)
        async_load_b128(ldsA0 + nb * bufA + j * rstep,
                        gA + (size_t)(j * 64) * KDIM + koff);
      #pragma unroll
      for (int j = 0; j < 2; ++j)
        async_load_b128(ldsB0 + nb * bufB + j * rstep,
                        gB + (size_t)(j * 64) * KDIM + koff);
    }

    const half_t* tA = sA[cur];
    const half_t* tB = sB[cur];
    const int hi = lane >> 4;        // 0: lanes 0-15, 1: lanes 16-31
    const int r  = lane & 15;

    // A fragments (16x32 f16): lanes<16 hold K 0-7 & 16-23, lanes>=16 K 8-15 & 24-31
    v16h af[4];
    #pragma unroll
    for (int mi = 0; mi < 4; ++mi) {
      const half_t* p = tA + (wm * 64 + mi * 16 + r) * LDT + hi * 8;
      v8h lo = *(const v8h*)p;
      v8h h2 = *(const v8h*)(p + 16);
      af[mi] = __builtin_shufflevector(lo, h2, 0,1,2,3,4,5,6,7,8,9,10,11,12,13,14,15);
    }
    // B fragments (32x16 f16): lanes<16 hold K 0-7 & 16-23 (N=lane), lanes>=16 K 8-15 & 24-31
    v16h bf[4];
    #pragma unroll
    for (int ni = 0; ni < 4; ++ni) {
      const half_t* p = tB + (wn * 64 + ni * 16 + r) * LDT + hi * 8;
      v8h lo = *(const v8h*)p;
      v8h h2 = *(const v8h*)(p + 16);
      bf[ni] = __builtin_shufflevector(lo, h2, 0,1,2,3,4,5,6,7,8,9,10,11,12,13,14,15);
    }

    #pragma unroll
    for (int mi = 0; mi < 4; ++mi)
      #pragma unroll
      for (int ni = 0; ni < 4; ++ni)
        acc[mi][ni] = __builtin_amdgcn_wmma_f32_16x16x32_f16(
            false, af[mi], false, bf[ni], (short)0, acc[mi][ni], false, false);

    wait_async0();                   // our async writes to next buffer done
    __syncthreads();                 // publish to all waves
  }

  // Epilogue: C/D layout — VGPR v: lanes<16 -> row v, lanes>=16 -> row v+8
  #pragma unroll
  for (int mi = 0; mi < 4; ++mi) {
    const int r0 = bm + wm * 64 + mi * 16 + ((lane >> 4) << 3);
    #pragma unroll
    for (int ni = 0; ni < 4; ++ni) {
      const int col = bn + wn * 64 + ni * 16 + (lane & 15);
      const float bv = bias[col];
      #pragma unroll
      for (int v = 0; v < 8; ++v)
        C[(size_t)(r0 + v) * OUT_F + col] = acc[mi][ni][v] + bv;
    }
  }
}

// ---------------------------------------------------------------------------
extern "C" void kernel_launch(void* const* d_in, const int* in_sizes, int n_in,
                              void* d_out, int out_size, void* d_ws, size_t ws_size,
                              hipStream_t stream) {
  const float* x    = (const float*)d_in[0];   // (rows, 1024)
  const float* w    = (const float*)d_in[1];   // (1024, 1024)
  const float* cheb = (const float*)d_in[2];   // (1024, 1024, 11)
  const float* bias = (const float*)d_in[3];   // (1024,)
  float* out = (float*)d_out;

  const int total = in_sizes[0];               // rows * 1024
  const int rows  = total / IN_F;              // 8192

  half_t* F = (half_t*)d_ws;                       // rows * 12288 f16
  half_t* G = F + (size_t)rows * KDIM;             // 1024 * 12288 f16

  cheby_expand_kernel<<<(total + 255) / 256, 256, 0, stream>>>(x, F, total);
  cheby_pack_weights_kernel<<<(OUT_F * IN_F) / 256, 256, 0, stream>>>(w, cheb, G);

  dim3 grid(rows / BM, OUT_F / BN);            // (32, 8) at rows=8192
  cheby_gemm_kernel<<<grid, 256, 0, stream>>>(F, G, bias, out, rows);
}
